// SuperpixelAttention_37185826849302
// MI455X (gfx1250) — matrix operand order repaired and checked by reading, not compiled, not run on hardware
//
#include <hip/hip_runtime.h>
#include <hip/hip_bf16.h>

// ---------------------------------------------------------------------------
// Superpixel local attention, MI455X (gfx1250, wave32, WMMA + async-to-LDS).
// HBM-bound (~1 GB, 23.3 TB/s => ~45us floor). GEMMs: v_wmma_f32_16x16x32_f16.
// Window math: the 25x neighbor reuse of sims/k is served from LDS via
// GLOBAL_LOAD_ASYNC_TO_LDS_B128 (ASYNCcnt) instead of hammering L2 (13+ GB of
// L2 reads in the naive version vs ~0.9 GB staged).
// ---------------------------------------------------------------------------

#define B_   8
#define C_   128
#define H_   128
#define W_   128
#define NH_  4
#define HD_  32
#define K2_  25
#define SG2_ 256
#define HW_  (H_ * W_)
#define NPIX (B_ * H_ * W_)
#define LDT  136   // f16 LDS row pitch (272B: 16B-aligned, bank-friendly)

typedef __attribute__((ext_vector_type(16))) _Float16 v16h;
typedef __attribute__((ext_vector_type(8)))  _Float16 v8h;
typedef __attribute__((ext_vector_type(8)))  float    v8f;
typedef __attribute__((ext_vector_type(4)))  float    v4f;

__device__ __forceinline__ v16h cat8(v8h a, v8h b) {
  return __builtin_shufflevector(a, b, 0,1,2,3,4,5,6,7,8,9,10,11,12,13,14,15);
}

// Async global->LDS 16B copy (per-lane; tracked by ASYNCcnt).
__device__ __forceinline__ void async_copy_b128(unsigned lds_off, const void* gptr) {
  asm volatile("global_load_async_to_lds_b128 %0, %1, off"
               :: "v"(lds_off), "v"(gptr) : "memory");
}
__device__ __forceinline__ void wait_async0() {
  asm volatile("s_wait_asynccnt 0" ::: "memory");
}

// A fragment (16x32 f16): lane<16 -> row m=lane, K {kt+0..7, kt+16..23};
// lane>=16 -> row m=lane-16, K {kt+8..15, kt+24..31}.  (ISA 7.12.2)
__device__ __forceinline__ v16h load_a_frag(const _Float16* As, int mbase, int kt, int lane) {
  int lm = lane & 15;
  int hi = lane >> 4;
  const _Float16* rp = As + (mbase + lm) * LDT + kt + hi * 8;
  v8h lo = *(const v8h*)(rp);
  v8h hh = *(const v8h*)(rp + 16);
  return cat8(lo, hh);
}

// B fragment (32x16 f16), LDS holds B transposed [n][k]:
// lane<16 -> col n=lane, K {kt+0..7, kt+16..23}; lane>=16 -> col n=lane-16, K {kt+8..15, kt+24..31}.
__device__ __forceinline__ v16h load_b_frag(const _Float16* Bs, int nbase, int kt, int lane) {
  int lm = lane & 15;
  int hi = lane >> 4;
  const _Float16* rp = Bs + (nbase + lm) * LDT + kt + hi * 8;
  v8h lo = *(const v8h*)(rp);
  v8h hh = *(const v8h*)(rp + 16);
  return cat8(lo, hh);
}

// ---------------------------------------------------------------------------
// Kernel 1: qk / v GEMM.  M tile = 128 pixels (one (b,h) row), N tile = 128.
// gridDim.y: 0 -> q cols, 1 -> k cols, 2 -> v cols.
// ---------------------------------------------------------------------------
__global__ void qkv_gemm_kernel(const float* __restrict__ x,
                                const float* __restrict__ w_qk,
                                const float* __restrict__ b_qk,
                                const float* __restrict__ w_v,
                                const float* __restrict__ b_v,
                                _Float16* __restrict__ qk,
                                _Float16* __restrict__ vout) {
  extern __shared__ char smem[];
  _Float16* As = (_Float16*)smem;                       // 128 x LDT
  _Float16* Bs = (_Float16*)(smem + 128 * LDT * 2);     // 128 x LDT, [n][k]

  const int tid   = threadIdx.x;
  const int mb    = blockIdx.x;          // 0..1023
  const int nb    = blockIdx.y;          // 0..2
  const int pbase = mb * 128;
  const int b     = pbase / HW_;
  const int hh    = (pbase % HW_) / W_;

  // A tile: As[w][c] = x[b][c][hh][w]  (global reads coalesced along w)
  const float* xrow = x + (size_t)b * C_ * HW_ + (size_t)hh * W_;
  #pragma unroll 4
  for (int i = 0; i < 64; ++i) {
    int idx = tid + i * 256;
    int c = idx >> 7, w = idx & 127;
    As[w * LDT + c] = (_Float16)xrow[(size_t)c * HW_ + w];
  }
  // B tile transposed: Bs[n][k] = W[k][n]
  #pragma unroll 4
  for (int i = 0; i < 64; ++i) {
    int idx = tid + i * 256;
    int n = idx >> 7, k = idx & 127;
    float wv = (nb < 2) ? w_qk[k * 256 + nb * 128 + n] : w_v[k * 128 + n];
    Bs[n * LDT + k] = (_Float16)wv;
  }
  __syncthreads();

  const int wave = tid >> 5, lane = tid & 31;
  const int Mb = (wave >> 1) * 32;
  const int Nb = (wave & 1) * 64;

  v8f acc[2][4] = {};
  #pragma unroll
  for (int kt = 0; kt < 128; kt += 32) {
    v16h a0 = load_a_frag(As, Mb,      kt, lane);
    v16h a1 = load_a_frag(As, Mb + 16, kt, lane);
    #pragma unroll
    for (int j = 0; j < 4; ++j) {
      v16h bf = load_b_frag(Bs, Nb + j * 16, kt, lane);
      acc[0][j] = __builtin_amdgcn_wmma_f32_16x16x32_f16(false, a0, false, bf,
                                                         (short)0, acc[0][j], false, false);
      acc[1][j] = __builtin_amdgcn_wmma_f32_16x16x32_f16(false, a1, false, bf,
                                                         (short)0, acc[1][j], false, false);
    }
  }

  // D layout: n = lane%16 (+16j+Nb), m = r + 8*(lane>=16) (+16i+Mb)
  const int lm = lane & 15, hi = lane >> 4;
  #pragma unroll
  for (int i = 0; i < 2; ++i) {
    #pragma unroll
    for (int j = 0; j < 4; ++j) {
      int nloc = Nb + j * 16 + lm;
      float bias = (nb < 2) ? b_qk[nb * 128 + nloc] : b_v[nloc];
      #pragma unroll
      for (int r = 0; r < 8; ++r) {
        int mloc = Mb + i * 16 + hi * 8 + r;
        size_t pg = (size_t)(pbase + mloc);
        float val = acc[i][j][r] + bias;
        if (nb < 2) qk[pg * 256 + nb * 128 + nloc] = (_Float16)val;
        else        vout[pg * 128 + nloc]          = (_Float16)val;
      }
    }
  }
}

// ---------------------------------------------------------------------------
// Kernel 2: per 16x16 pixel tile, stage 20x20 halos of k (f16) and sims
// (f32, 4 chunks of 64) into LDS via async copies; then window scores,
// softmax, superpixel reweight entirely from LDS.
// ---------------------------------------------------------------------------
#define SPITCH 68              // f32 pitch for sims halo rows (272B)
__global__ void attn_kernel(const _Float16* __restrict__ qk,
                            const float* __restrict__ sims,
                            float* __restrict__ attn) {
  extern __shared__ char smem[];
  float*    Ssh = (float*)smem;                          // 400 x 68 f32 chunk
  _Float16* Ksh = (_Float16*)(smem + 400 * SPITCH * 4);  // 400 x LDT f16

  const int tid = threadIdx.x;
  const int bt  = blockIdx.x;
  const int b   = bt >> 6;
  const int th0 = ((bt >> 3) & 7) * 16;
  const int tw0 = (bt & 7) * 16;
  const int ty = tid >> 4, tx = tid & 15;
  const int hh = th0 + ty, ww = tw0 + tx;
  const int p  = (b * H_ + hh) * W_ + ww;

  // ---- issue async loads of the k halo (400 px x 128 f16 = 6400 x 16B) ----
  for (int i = 0; i < 25; ++i) {
    int flat = tid + i * 256;
    int idx = flat >> 4, vec = flat & 15;
    int r = idx / 20, c = idx % 20;
    int gi = th0 + r - 2, gj = tw0 + c - 2;
    if (gi >= 0 && gi < H_ && gj >= 0 && gj < W_) {
      int pg = (b * H_ + gi) * W_ + gj;
      unsigned loff = (unsigned)(uintptr_t)(Ksh + idx * LDT + vec * 8);
      async_copy_b128(loff, qk + (size_t)pg * 256 + 128 + vec * 8);
    }
  }

  // per-offset validity + halo index (matches reference zero padding)
  bool val[K2_];
  int  nidx[K2_];
  #pragma unroll
  for (int o = 0; o < K2_; ++o) {
    int di = o / 5, dj = o % 5;
    int ni = hh + di - 2, nj = ww + dj - 2;
    val[o]  = (ni >= 0 && ni < H_ && nj >= 0 && nj < W_);
    nidx[o] = (ty + di) * 20 + (tx + dj);
  }

  // ---- pwd: 4 async-staged chunks of 64 superpixels ----
  float pwd[K2_] = {};
  const int cidx = (ty + 2) * 20 + (tx + 2);
  for (int ch = 0; ch < 4; ++ch) {
    __syncthreads();                      // previous chunk fully consumed
    for (int i = 0; i < 25; ++i) {        // 400 px x 64 f32 = 6400 x 16B
      int flat = tid + i * 256;
      int idx = flat >> 4, vec = flat & 15;
      int r = idx / 20, c = idx % 20;
      int gi = th0 + r - 2, gj = tw0 + c - 2;
      if (gi >= 0 && gi < H_ && gj >= 0 && gj < W_) {
        int pg = (b * H_ + gi) * W_ + gj;
        unsigned loff = (unsigned)(uintptr_t)(Ssh + idx * SPITCH + vec * 4);
        async_copy_b128(loff, sims + (size_t)pg * SG2_ + ch * 64 + vec * 4);
      }
    }
    wait_async0();
    __syncthreads();
    const float* cen = Ssh + cidx * SPITCH;
    for (int s = 0; s < 64; s += 4) {
      v4f cv = *(const v4f*)(cen + s);
      #pragma unroll
      for (int o = 0; o < K2_; ++o) {
        if (val[o]) {
          v4f nv = *(const v4f*)(Ssh + nidx[o] * SPITCH + s);
          pwd[o] += cv[0] * nv[0] + cv[1] * nv[1] + cv[2] * nv[2] + cv[3] * nv[3];
        }
      }
    }
  }

  // ---- per-head scores (k from LDS) + softmax + reweight ----
  const _Float16* qp = qk + (size_t)p * 256;
  const float scale = 0.17677669529663687f;  // HD^-0.5
  for (int h = 0; h < NH_; ++h) {
    v8h q0 = *(const v8h*)(qp + h * HD_ + 0);
    v8h q1 = *(const v8h*)(qp + h * HD_ + 8);
    v8h q2 = *(const v8h*)(qp + h * HD_ + 16);
    v8h q3 = *(const v8h*)(qp + h * HD_ + 24);

    float sc_[K2_];
    #pragma unroll
    for (int o = 0; o < K2_; ++o) {
      float d = 0.f;                       // OOB -> zero-padded k -> score 0
      if (val[o]) {
        const _Float16* kp = Ksh + nidx[o] * LDT + h * HD_;
        v8h k0 = *(const v8h*)(kp + 0);
        v8h k1 = *(const v8h*)(kp + 8);
        v8h k2 = *(const v8h*)(kp + 16);
        v8h k3 = *(const v8h*)(kp + 24);
        #pragma unroll
        for (int e = 0; e < 8; ++e) {
          d += (float)q0[e] * (float)k0[e];
          d += (float)q1[e] * (float)k1[e];
          d += (float)q2[e] * (float)k2[e];
          d += (float)q3[e] * (float)k3[e];
        }
      }
      sc_[o] = d * scale;
    }
    float mx = sc_[0];
    #pragma unroll
    for (int o = 1; o < K2_; ++o) mx = fmaxf(mx, sc_[o]);
    float sum = 0.f;
    #pragma unroll
    for (int o = 0; o < K2_; ++o) { float e = __expf(sc_[o] - mx); sc_[o] = e; sum += e; }
    float inv = 1.f / sum;
    float tot = 0.f;
    #pragma unroll
    for (int o = 0; o < K2_; ++o) { float w = sc_[o] * inv * pwd[o]; sc_[o] = w; tot += w; }
    float d2 = 1.f / fmaxf(tot, 1e-8f);
    float* ap = attn + ((size_t)p * NH_ + h) * K2_;
    #pragma unroll
    for (int o = 0; o < K2_; ++o) ap[o] = sc_[o] * d2;
  }
}

// ---------------------------------------------------------------------------
// Kernel 3: neighbor aggregation of V (LDS) fused with projection GEMM (WMMA),
// transposed coalesced store into B,C,H,W via LDS staging.
// ---------------------------------------------------------------------------
__global__ void agg_proj_kernel(const _Float16* __restrict__ vbuf,
                                const float* __restrict__ attn,
                                const float* __restrict__ w_proj,
                                const float* __restrict__ b_proj,
                                float* __restrict__ out) {
  extern __shared__ char smem[];
  _Float16* Wp     = (_Float16*)smem;                       // 128 x LDT  (34816 B)
  float*    AttnS  = (float*)(smem + 34816);                // 128 x 100  (51200 B)
  _Float16* Agg    = (_Float16*)(smem + 34816 + 51200);     // 128 x LDT  (34816 B)
  float*    Dst    = (float*)(smem + 34816);                // 128 x 129 f32, reused

  const int tid   = threadIdx.x;
  const int pbase = blockIdx.x * 128;
  const int b     = pbase / HW_;
  const int hh    = (pbase % HW_) / W_;

  // attn tile (12800 f32 = 3200 x 16B) via async path
  for (int i = 0; i < 13; ++i) {
    int flat = tid + i * 256;
    if (flat < 3200) {
      unsigned loff = (unsigned)(uintptr_t)(AttnS + flat * 4);
      async_copy_b128(loff, attn + (size_t)pbase * (NH_ * K2_) + flat * 4);
    }
  }
  #pragma unroll 4
  for (int i = 0; i < 64; ++i) {
    int idx = tid + i * 256;
    int n = idx >> 7, k = idx & 127;
    Wp[n * LDT + k] = (_Float16)w_proj[k * 128 + n];
  }
  wait_async0();
  __syncthreads();

  // Aggregation: thread owns (m = tid/2, 64 channels). Agg[m][c] f16.
  {
    const int m = tid >> 1;
    const int cbase = (tid & 1) * 64;
    float acc[64];
    #pragma unroll
    for (int cc = 0; cc < 64; ++cc) acc[cc] = 0.f;
    #pragma unroll
    for (int o = 0; o < K2_; ++o) {
      int di = o / 5 - 2, dj = o % 5 - 2;
      int ni = hh + di, nj = m + dj;
      if (ni >= 0 && ni < H_ && nj >= 0 && nj < W_) {
        const _Float16* vp = vbuf + (size_t)(b * HW_ + ni * W_ + nj) * 128 + cbase;
        float a0 = AttnS[m * 100 + ((cbase      ) >> 5) * K2_ + o];
        float a1 = AttnS[m * 100 + ((cbase + 32 ) >> 5) * K2_ + o];
        #pragma unroll
        for (int cc = 0; cc < 32; ++cc) acc[cc] += a0 * (float)vp[cc];
        #pragma unroll
        for (int cc = 32; cc < 64; ++cc) acc[cc] += a1 * (float)vp[cc];
      }
    }
    #pragma unroll
    for (int cc = 0; cc < 64; ++cc)
      Agg[m * LDT + cbase + cc] = (_Float16)acc[cc];
  }
  __syncthreads();

  // Projection GEMM: Agg(128x128) x Wp^T + b_proj
  const int wave = tid >> 5, lane = tid & 31;
  const int Mb = (wave >> 1) * 32;
  const int Nb = (wave & 1) * 64;
  v8f acc[2][4] = {};
  #pragma unroll
  for (int kt = 0; kt < 128; kt += 32) {
    v16h a0 = load_a_frag(Agg, Mb,      kt, lane);
    v16h a1 = load_a_frag(Agg, Mb + 16, kt, lane);
    #pragma unroll
    for (int j = 0; j < 4; ++j) {
      v16h bf = load_b_frag(Wp, Nb + j * 16, kt, lane);
      acc[0][j] = __builtin_amdgcn_wmma_f32_16x16x32_f16(false, a0, false, bf,
                                                         (short)0, acc[0][j], false, false);
      acc[1][j] = __builtin_amdgcn_wmma_f32_16x16x32_f16(false, a1, false, bf,
                                                         (short)0, acc[1][j], false, false);
    }
  }
  __syncthreads();   // done reading Agg/AttnS; reuse region for Dst

  const int lm = lane & 15, hi = lane >> 4;
  #pragma unroll
  for (int i = 0; i < 2; ++i) {
    #pragma unroll
    for (int j = 0; j < 4; ++j) {
      int nloc = Nb + j * 16 + lm;
      float bias = b_proj[nloc];
      #pragma unroll
      for (int r = 0; r < 8; ++r) {
        int mloc = Mb + i * 16 + hi * 8 + r;
        Dst[nloc * 129 + mloc] = acc[i][j][r] + bias;   // pitch 129: conflict-free
      }
    }
  }
  __syncthreads();

  // Coalesced transposed store: out[b][n][hh][m]
  #pragma unroll 4
  for (int i = 0; i < 64; ++i) {
    int idx = tid + i * 256;
    int n = idx >> 7, m = idx & 127;
    out[(((size_t)b * C_ + n) * H_ + hh) * W_ + m] = Dst[n * 129 + m];
  }
}

// ---------------------------------------------------------------------------
extern "C" void kernel_launch(void* const* d_in, const int* in_sizes, int n_in,
                              void* d_out, int out_size, void* d_ws, size_t ws_size,
                              hipStream_t stream) {
  (void)in_sizes; (void)n_in; (void)out_size; (void)ws_size;
  const float* x      = (const float*)d_in[0];
  const float* sims   = (const float*)d_in[1];
  const float* w_qk   = (const float*)d_in[2];
  const float* b_qk   = (const float*)d_in[3];
  const float* w_v    = (const float*)d_in[4];
  const float* b_v    = (const float*)d_in[5];
  const float* w_proj = (const float*)d_in[6];
  const float* b_proj = (const float*)d_in[7];
  float* out = (float*)d_out;

  // Workspace: qk f16 (NPIX*256), v f16 (NPIX*128), attn f32 (NPIX*100) ~146MB
  _Float16* qk_ws = (_Float16*)d_ws;
  _Float16* v_ws  = (_Float16*)((char*)d_ws + (size_t)NPIX * 256 * 2);
  float*    a_ws  = (float*)((char*)d_ws + (size_t)NPIX * 256 * 2 + (size_t)NPIX * 128 * 2);

  const int mblocks = NPIX / 128;               // 1024
  qkv_gemm_kernel<<<dim3(mblocks, 3), 256, 2 * 128 * LDT * 2, stream>>>(
      x, w_qk, b_qk, w_v, b_v, qk_ws, v_ws);
  attn_kernel<<<NPIX / 256, 256, 400 * SPITCH * 4 + 400 * LDT * 2, stream>>>(
      qk_ws, sims, a_ws);
  agg_proj_kernel<<<mblocks, 256, 34816 + 51200 + 34816, stream>>>(
      v_ws, a_ws, w_proj, b_proj, out);
}
